// MulHeadAttention_10823317586454
// MI455X (gfx1250) — matrix operand order repaired and checked by reading, not compiled, not run on hardware
//
#include <hip/hip_runtime.h>

typedef __attribute__((ext_vector_type(16))) _Float16 v16h;
typedef __attribute__((ext_vector_type(8)))  _Float16 v8h;
typedef __attribute__((ext_vector_type(8)))  float    v8f;

#define N_TOK 4096
#define DIM   512

static __device__ __forceinline__ v16h hcombine(v8h a, v8h b) {
    return __builtin_shufflevector(a, b, 0,1,2,3,4,5,6,7,8,9,10,11,12,13,14,15);
}

// A-matrix fragment (16x32 f16, M x K). base points at row 0; ldk = row pitch in halves.
// Lane L: M = L&15, h = L>>4; elems i = K[h*16 + i] (contiguous 32 bytes).
static __device__ __forceinline__ v16h load_frag_a(const _Float16* base, int ldk,
                                                   int rowbase, int k0, int lane) {
    int m = lane & 15, h = lane >> 4;
    const _Float16* p = base + (size_t)(rowbase + m) * ldk + k0 + h * 16;
    v8h c0 = *(const v8h*)(p);
    v8h c1 = *(const v8h*)(p + 8);
    return hcombine(c0, c1);
}

// B-matrix fragment (32x16 f16, K x N), sourced from memory laid out as [n][k] (k contiguous).
// Lane L: N = L&15, h = L>>4; elems i = K[h*16 + i] (contiguous 32 bytes).
static __device__ __forceinline__ v16h load_frag_b(const _Float16* base, int ldk,
                                                   int nbase, int k0, int lane) {
    int n = lane & 15, h = lane >> 4;
    const _Float16* p = base + (size_t)(nbase + n) * ldk + k0 + h * 16;
    v8h c0 = *(const v8h*)(p);
    v8h c1 = *(const v8h*)(p + 8);
    return hcombine(c0, c1);
}

static __device__ __forceinline__ v8f wmma_f16(v16h a, v16h b, v8f c) {
    return __builtin_amdgcn_wmma_f32_16x16x32_f16(false, a, false, b, (short)0, c, false, false);
}

// ---------------------------------------------------------------------------
// Kernel 1: q/k/v projections. grid = (4096/32, 512/128, 3), block = 256.
// P = X @ W + b in f16. proj 0 -> q (row major), 1 -> k (row major),
// 2 -> v stored transposed as vT[DIM][N_TOK].
// ---------------------------------------------------------------------------
__global__ __launch_bounds__(256) void proj_kernel(
    const float* __restrict__ X,
    const float* __restrict__ Wq, const float* __restrict__ bq,
    const float* __restrict__ Wk, const float* __restrict__ bk,
    const float* __restrict__ Wv, const float* __restrict__ bv,
    _Float16* __restrict__ qf, _Float16* __restrict__ kf, _Float16* __restrict__ vt)
{
    __shared__ alignas(16) _Float16 Xs[32][32];    // [m][k]
    __shared__ alignas(16) _Float16 Ws[128][32];   // [n][k]  (W tile transposed)

    const int tid  = threadIdx.x;
    const int lane = tid & 31;
    const int wave = tid >> 5;          // 0..7
    const int wm   = wave >> 2;         // 0..1  (M sub-tile)
    const int wn   = wave & 3;          // 0..3  (N sub-tile, 32 cols each)

    const int row0 = blockIdx.x * 32;
    const int col0 = blockIdx.y * 128;
    const int proj = blockIdx.z;        // 0..2

    const float* W    = (proj == 0) ? Wq : (proj == 1) ? Wk : Wv;
    const float* bias = (proj == 0) ? bq : (proj == 1) ? bk : bv;

    v8f acc0 = {}, acc1 = {};

    for (int k0 = 0; k0 < DIM; k0 += 32) {
        // Stage X tile (32x32): 1024 f32, one float4 per thread.
        {
            int m  = tid >> 3;
            int kk = (tid & 7) * 4;
            float4 x = *(const float4*)(X + (size_t)(row0 + m) * DIM + k0 + kk);
            Xs[m][kk + 0] = (_Float16)x.x;
            Xs[m][kk + 1] = (_Float16)x.y;
            Xs[m][kk + 2] = (_Float16)x.z;
            Xs[m][kk + 3] = (_Float16)x.w;
        }
        // Stage W tile (K=32 x N=128) transposed into Ws[n][k]: 4096 f32.
        #pragma unroll
        for (int j = 0; j < 4; ++j) {
            int linear = tid + j * 256;          // 0..1023 float4s
            int kk = linear >> 5;                // 0..31
            int n4 = (linear & 31) * 4;          // 0..124
            float4 w = *(const float4*)(W + (size_t)(k0 + kk) * DIM + col0 + n4);
            Ws[n4 + 0][kk] = (_Float16)w.x;
            Ws[n4 + 1][kk] = (_Float16)w.y;
            Ws[n4 + 2][kk] = (_Float16)w.z;
            Ws[n4 + 3][kk] = (_Float16)w.w;
        }
        __syncthreads();

        v16h a  = load_frag_a(&Xs[0][0], 32, wm * 16, 0, lane);
        v16h b0 = load_frag_b(&Ws[0][0], 32, wn * 32,      0, lane);
        acc0 = wmma_f16(a, b0, acc0);
        v16h b1 = load_frag_b(&Ws[0][0], 32, wn * 32 + 16, 0, lane);
        acc1 = wmma_f16(a, b1, acc1);
        __syncthreads();
    }

    // Bias add + store (f16). Address pattern is uniform per block (proj).
    _Float16* dst = (proj == 0) ? qf : (proj == 1) ? kf : vt;
    const bool tr = (proj == 2);
    const int h  = lane >> 4;
    const int nc = lane & 15;
    #pragma unroll
    for (int t = 0; t < 2; ++t) {
        v8f acc = t ? acc1 : acc0;
        int col = col0 + wn * 32 + t * 16 + nc;
        float bval = bias[col];
        #pragma unroll
        for (int r = 0; r < 8; ++r) {
            int row = row0 + wm * 16 + r + h * 8;
            float val = acc[r] + bval;
            size_t idx = tr ? ((size_t)col * N_TOK + row)
                            : ((size_t)row * DIM + col);
            dst[idx] = (_Float16)val;
        }
    }
}

// ---------------------------------------------------------------------------
// Kernel 2: attention. grid = 4096/32 = 128 blocks, block = 512 threads (16 waves).
// Each block owns a 32-row query strip (uses CDNA5's 320 KB WGP LDS: ~290 KB):
//   S = q@k^T (f16 in LDS), row softmax, O = P@v, f32 out.
// B-fragments stream straight from global (row-major k / transposed vT) —
// each lane loads 16 contiguous k of its column (k = 16h + i, as A), so no staging/transpose needed.
// ---------------------------------------------------------------------------
__global__ __launch_bounds__(512) void attn_kernel(
    const _Float16* __restrict__ qf,
    const _Float16* __restrict__ kf,
    const _Float16* __restrict__ vt,
    float* __restrict__ out)
{
    __shared__ alignas(16) _Float16 S[32 * N_TOK];   // 256 KB score/prob strip
    __shared__ alignas(16) _Float16 qs[32 * DIM];    // 32 KB query strip
    __shared__ float red[32][16];                    // softmax reduction scratch

    const int tid  = threadIdx.x;
    const int lane = tid & 31;
    const int wave = tid >> 5;          // 0..15
    const int row0 = blockIdx.x * 32;
    const int h    = lane >> 4;
    const int nc   = lane & 15;

    // Stage the 32x512 query strip (f16): 2048 int4s, 4 per thread.
    {
        const int4* src = (const int4*)(qf + (size_t)row0 * DIM);
        int4* dst = (int4*)qs;
        #pragma unroll
        for (int i = 0; i < 4; ++i) dst[tid + i * 512] = src[tid + i * 512];
    }
    __syncthreads();

    // ---------------- Phase 1: S = q @ k^T ----------------
    // Wave (wm, wj): m-strip wm*16, two 16-col j tiles at jb + wj*32.
    // Each A fragment (ds_load) feeds two WMMAs; K-loop unrolled for ILP.
    {
        const int wm = wave >> 3;       // 0..1
        const int wj = wave & 7;        // 0..7
        for (int jb = 0; jb < N_TOK; jb += 256) {
            const _Float16* krow0 = kf + (size_t)(jb + wj * 32 + nc) * DIM + h * 16;
            const _Float16* krow1 = krow0 + (size_t)16 * DIM;
            if (jb + 256 < N_TOK) {     // prefetch next j-stripe's k rows
                __builtin_prefetch(krow0 + (size_t)256 * DIM, 0, 1);
                __builtin_prefetch(krow1 + (size_t)256 * DIM, 0, 1);
            }
            v8f acc0 = {}, acc1 = {};
            #pragma unroll
            for (int k0 = 0; k0 < DIM; k0 += 32) {
                v16h a  = load_frag_a(qs, DIM, wm * 16, k0, lane);
                v16h b0 = hcombine(*(const v8h*)(krow0 + k0),
                                   *(const v8h*)(krow0 + k0 + 8));
                v16h b1 = hcombine(*(const v8h*)(krow1 + k0),
                                   *(const v8h*)(krow1 + k0 + 8));
                acc0 = wmma_f16(a, b0, acc0);
                acc1 = wmma_f16(a, b1, acc1);
            }
            int jc = jb + wj * 32 + nc;
            #pragma unroll
            for (int r = 0; r < 8; ++r) {
                S[(size_t)(wm * 16 + r + h * 8) * N_TOK + jc]      = (_Float16)acc0[r];
                S[(size_t)(wm * 16 + r + h * 8) * N_TOK + jc + 16] = (_Float16)acc1[r];
            }
        }
    }
    __syncthreads();

    // ---------------- Phase 2: row softmax (vectorized b128 LDS) ----------------
    {
        const int row  = tid >> 4;       // 0..31
        const int tcol = tid & 15;       // 16 threads/row, contiguous 256-elem chunks
        _Float16* Srow = S + (size_t)row * N_TOK + tcol * 256;

        float pm = -3.0e38f;
        #pragma unroll 4
        for (int j = 0; j < 256; j += 8) {
            v8h x = *(const v8h*)(Srow + j);
            #pragma unroll
            for (int e = 0; e < 8; ++e) pm = fmaxf(pm, (float)x[e]);
        }
        red[row][tcol] = pm;
        __syncthreads();
        if (tcol == 0) {
            float m2 = red[row][0];
            #pragma unroll
            for (int t = 1; t < 16; ++t) m2 = fmaxf(m2, red[row][t]);
            red[row][0] = m2;
        }
        __syncthreads();
        const float mrow = red[row][0];
        __syncthreads();

        float ps = 0.0f;
        #pragma unroll 4
        for (int j = 0; j < 256; j += 8) {
            v8h x = *(const v8h*)(Srow + j);
            v8h y;
            #pragma unroll
            for (int e = 0; e < 8; ++e) {
                float ef = __expf((float)x[e] - mrow);
                ps += ef;
                y[e] = (_Float16)ef;
            }
            *(v8h*)(Srow + j) = y;
        }
        red[row][tcol] = ps;
        __syncthreads();
        if (tcol == 0) {
            float s2 = 0.0f;
            #pragma unroll
            for (int t = 0; t < 16; ++t) s2 += red[row][t];
            red[row][0] = s2;
        }
        __syncthreads();
        const float inv = 1.0f / red[row][0];
        #pragma unroll 4
        for (int j = 0; j < 256; j += 8) {
            v8h x = *(const v8h*)(Srow + j);
            #pragma unroll
            for (int e = 0; e < 8; ++e) x[e] = (_Float16)((float)x[e] * inv);
            *(v8h*)(Srow + j) = x;
        }
        __syncthreads();
    }

    // ---------------- Phase 3: O = P @ v ----------------
    // Wave (wm, wn): m-strip wm*16, output columns wn*64..+63 (4 n-tiles of 16).
    {
        const int wm = wave >> 3;       // 0..1
        const int wn = wave & 7;        // 0..7
        v8f oacc[4] = {{}, {}, {}, {}};
        const _Float16* vrow0 = vt + (size_t)(wn * 64 + nc) * N_TOK + h * 16;
        #pragma unroll 4
        for (int j0 = 0; j0 < N_TOK; j0 += 32) {
            v16h a = load_frag_a(S, N_TOK, wm * 16, j0, lane);  // probs (f16, LDS)
            #pragma unroll
            for (int t = 0; t < 4; ++t) {
                const _Float16* p = vrow0 + (size_t)(t * 16) * N_TOK + j0;
                if (j0 + 32 < N_TOK)
                    __builtin_prefetch(p + 32, 0, 1);
                v16h b = hcombine(*(const v8h*)(p), *(const v8h*)(p + 8));
                oacc[t] = wmma_f16(a, b, oacc[t]);
            }
        }
        #pragma unroll
        for (int t = 0; t < 4; ++t) {
            int col = wn * 64 + t * 16 + nc;
            #pragma unroll
            for (int r = 0; r < 8; ++r)
                out[(size_t)(row0 + wm * 16 + r + h * 8) * DIM + col] = oacc[t][r];
        }
    }
}

extern "C" void kernel_launch(void* const* d_in, const int* in_sizes, int n_in,
                              void* d_out, int out_size, void* d_ws, size_t ws_size,
                              hipStream_t stream) {
    const float* X  = (const float*)d_in[0];
    const float* Wq = (const float*)d_in[1];
    const float* bq = (const float*)d_in[2];
    const float* Wk = (const float*)d_in[3];
    const float* bk = (const float*)d_in[4];
    const float* Wv = (const float*)d_in[5];
    const float* bv = (const float*)d_in[6];
    float* out = (float*)d_out;

    _Float16* qf = (_Float16*)d_ws;                      // [4096][512] f16
    _Float16* kf = qf + (size_t)N_TOK * DIM;             // [4096][512] f16
    _Float16* vt = kf + (size_t)N_TOK * DIM;             // [512][4096] f16 (v transposed)

    proj_kernel<<<dim3(N_TOK / 32, DIM / 128, 3), 256, 0, stream>>>(
        X, Wq, bq, Wk, bk, Wv, bv, qf, kf, vt);
    attn_kernel<<<dim3(N_TOK / 32), 512, 0, stream>>>(qf, kf, vt, out);
}
